// Predictor_65197603554257
// MI455X (gfx1250) — matrix operand order repaired and checked by reading, not compiled, not run on hardware
//
#include <hip/hip_runtime.h>
#include <hip/hip_bf16.h>

typedef __attribute__((ext_vector_type(16))) _Float16 v16h;
typedef __attribute__((ext_vector_type(8)))  _Float16 v8h;
typedef __attribute__((ext_vector_type(8)))  float    v8f;

#define MAXF   100
#define BN     16
#define TT     512
#define TMF    2000
#define DTXT   256
#define DIN    660
#define DINP   672      // padded to 21*32
#define HD     512
#define NG     1536     // 3*H
#define D2     1024     // 2*H
#define HPAD   8

// ---------------------------------------------------------------------------
// fragment helpers (CDNA5 WMMA 16x16x32 f16 layouts per ISA 7.12.2)
// ---------------------------------------------------------------------------
__device__ __forceinline__ v16h cat8(v8h lo, v8h hi) {
  return __builtin_shufflevector(lo, hi, 0,1,2,3,4,5,6,7,8,9,10,11,12,13,14,15);
}

// A (16xK tile, row-major, rows=M): lanes 0-15 & 16-31 both M=lane&15.
// lanes 0-15: VGPR0-3 -> K 0..7, VGPR4-7 -> K 16..23 ; lanes 16-31: +8
__device__ __forceinline__ v16h loadA(const _Float16* __restrict__ base, int ld, int lane) {
  int m = lane & 15, half = lane >> 4;
  const _Float16* p = base + (size_t)m * ld + 8 * half;
  v8h lo = *(const v8h*)(p);
  v8h hi = *(const v8h*)(p + 16);
  return cat8(lo, hi);
}

// B (Kx16 tile) loaded from N-major weights W[n][k] (so W row = B column).
// lanes 0-15: K 0..7 then 16..23, lanes 16-31: +8 (same as A).
__device__ __forceinline__ v16h loadB(const _Float16* __restrict__ base, int ld, int lane) {
  int n = lane & 15, half = lane >> 4;
  const _Float16* p = base + (size_t)n * ld + 8 * half;
  v8h lo = *(const v8h*)(p);
  v8h hi = *(const v8h*)(p + 16);
  return cat8(lo, hi);
}

// ---------------------------------------------------------------------------
// f32 -> f16 weight conversion with K padding
// ---------------------------------------------------------------------------
__global__ __launch_bounds__(256) void cvt_pad_kernel(
    const float* __restrict__ src, _Float16* __restrict__ dst,
    int rows, int scols, int dcols)
{
  int idx = blockIdx.x * 256 + threadIdx.x;
  if (idx >= rows * dcols) return;
  int r = idx / dcols, c = idx - r * dcols;
  dst[idx] = (c < scols) ? (_Float16)src[(size_t)r * scols + c] : (_Float16)0.f;
}

// ---------------------------------------------------------------------------
// clip_score feature build: X[t][b][672] f16 = [texts(256) | c1(202) | c2(202) | pad]
// ---------------------------------------------------------------------------
__global__ __launch_bounds__(256) void feat_kernel(
    const float* __restrict__ texts, const float* __restrict__ w1,
    const float* __restrict__ w2, const int* __restrict__ tlen,
    const int* __restrict__ mlen, _Float16* __restrict__ X)
{
  int t = blockIdx.x;          // 0..511
  int b = blockIdx.y;          // 0..15
  int L = tlen[b], M = mlen[b];
  bool tvalid = t < L;
  float mf = 100.0f + (float)t * ((float)(M - 1) / (float)(L - 1));
  int mid = (int)mf;           // truncation, matches astype(int32)
  int top = mid + MAXF, bot = mid - MAXF;
  _Float16* xr = X + ((size_t)t * BN + b) * DINP;
  for (int c = threadIdx.x; c < DINP; c += blockDim.x) {
    float v = 0.0f;
    if (c < DTXT) {
      v = texts[((size_t)b * TT + t) * DTXT + c];
    } else if (c < DTXT + 202) {
      int j = c - DTXT;
      if (tvalid) {
        if (j < 2 * MAXF) {
          int orig = mid - 2 * MAXF + j;
          if (orig >= 0 && orig < TMF && orig < M)
            v = w1[((size_t)b * TMF + orig) * TT + t];   // w1 swapaxes(1,2)
        } else v = (j == 2 * MAXF) ? (float)top : (float)bot;
      }
    } else if (c < DIN) {
      int j = c - (DTXT + 202);
      if (tvalid) {
        if (j < 2 * MAXF) {
          int orig = mid - 2 * MAXF + j;
          if (orig >= 0 && orig < TMF && orig < M)
            v = w2[((size_t)b * TT + t) * TMF + orig];
        } else v = (j == 2 * MAXF) ? (float)top : (float)bot;
      }
    }
    xr[c] = (_Float16)v;
  }
}

// ---------------------------------------------------------------------------
// Batched input projection: C[mt*16+m][n] = X @ W^T + bias  (W is [NG][ldx] f16)
// one wave per 16x16 output tile
// ---------------------------------------------------------------------------
__global__ __launch_bounds__(256) void gemm_xw_kernel(
    const _Float16* __restrict__ X, int ldx,
    const _Float16* __restrict__ W,
    const float* __restrict__ bias,
    float* __restrict__ C,
    int Ktiles, int Mtiles)
{
  int lane = threadIdx.x & 31;
  int wave = threadIdx.x >> 5;
  int g = blockIdx.x * 8 + wave;
  if (g >= Mtiles * (NG / 16)) return;
  int mt = g / (NG / 16);
  int nt = g - mt * (NG / 16);
  const _Float16* Xb = X + (size_t)mt * 16 * ldx;
  const _Float16* Wb = W + (size_t)nt * 16 * ldx;
  v8f acc = {};
  for (int k = 0; k < Ktiles; ++k) {
    v16h a = loadA(Xb + k * 32, ldx, lane);
    v16h b = loadB(Wb + k * 32, ldx, lane);
    acc = __builtin_amdgcn_wmma_f32_16x16x32_f16(false, a, false, b,
                                                 (short)0, acc, false, false);
  }
  int half = lane >> 4;
  int col = nt * 16 + (lane & 15);
  float bs = bias[col];
  float* Cb = C + (size_t)mt * 16 * NG + col;
#pragma unroll
  for (int r = 0; r < 8; ++r)
    Cb[(size_t)(r + 8 * half) * NG] = acc[r] + bs;
}

// ---------------------------------------------------------------------------
// Persistent GRU scan: grid = {fwd, bwd}, 32 waves, wave w owns h cols [16w,16w+16)
// h state in LDS (f16 for WMMA A, f32 for the blend). W_hh streamed (L2-resident).
// ---------------------------------------------------------------------------
__global__ __launch_bounds__(1024) void gru_scan_kernel(
    const float* __restrict__ xWf, const float* __restrict__ xWb,
    const _Float16* __restrict__ whhF, const _Float16* __restrict__ whhB,
    const float* __restrict__ bhhF, const float* __restrict__ bhhB,
    const int* __restrict__ tlen,
    float* __restrict__ outF32,        // [TT][16][1024]
    _Float16* __restrict__ outF16)     // same layout (f16 for next-layer GEMM) or null
{
  __shared__ __align__(16) _Float16 hf16[16][HD + HPAD];
  __shared__ float hf32[16][HD];
  __shared__ int lens[16];

  int dir = blockIdx.x;
  const float* xW       = dir ? xWb  : xWf;
  const _Float16* whh   = dir ? whhB : whhF;
  const float* bhh      = dir ? bhhB : bhhF;

  int lane = threadIdx.x & 31;
  int wave = threadIdx.x >> 5;       // 0..31
  int half = lane >> 4;
  int col  = lane & 15;
  int n0   = wave * 16;

  if (threadIdx.x < 16) lens[threadIdx.x] = tlen[threadIdx.x];
  for (int i = threadIdx.x; i < 16 * (HD + HPAD); i += blockDim.x)
    (&hf16[0][0])[i] = (_Float16)0.f;
  for (int i = threadIdx.x; i < 16 * HD; i += blockDim.x)
    (&hf32[0][0])[i] = 0.0f;
  __syncthreads();

  float br = bhh[0 * HD + n0 + col];
  float bz = bhh[1 * HD + n0 + col];
  float bn = bhh[2 * HD + n0 + col];

  int lrow[8];
#pragma unroll
  for (int r = 0; r < 8; ++r) lrow[r] = lens[r + 8 * half];

  const _Float16* arow = &hf16[col][0];       // A rows: m = lane&15

  for (int s = 0; s < TT; ++s) {
    // xW fragments for this step (per-row gathered time index for bwd dir)
    float xn8[8];
    v8f accR = {}, accZ = {}, accN = {};
#pragma unroll
    for (int r = 0; r < 8; ++r) {
      int m = r + 8 * half;
      int L = lrow[r];
      int t = dir ? ((s < L) ? (L - 1 - s) : s) : s;
      const float* row = xW + ((size_t)t * 16 + m) * NG;
      accR[r] = row[0 * HD + n0 + col];
      accZ[r] = row[1 * HD + n0 + col];
      xn8[r]  = row[2 * HD + n0 + col];
    }
    // recurrent matmul: 3 gate tiles share the A fragment
#pragma unroll 4
    for (int kk = 0; kk < HD / 32; ++kk) {
      int k0 = kk * 32;
      v8h alo = *(const v8h*)(arow + k0 + 8 * half);
      v8h ahi = *(const v8h*)(arow + k0 + 8 * half + 16);
      v16h a = cat8(alo, ahi);
      v16h bR = loadB(whh + (size_t)(0 * HD + n0) * HD + k0, HD, lane);
      v16h bZ = loadB(whh + (size_t)(1 * HD + n0) * HD + k0, HD, lane);
      v16h bN = loadB(whh + (size_t)(2 * HD + n0) * HD + k0, HD, lane);
      accR = __builtin_amdgcn_wmma_f32_16x16x32_f16(false, a, false, bR, (short)0, accR, false, false);
      accZ = __builtin_amdgcn_wmma_f32_16x16x32_f16(false, a, false, bZ, (short)0, accZ, false, false);
      accN = __builtin_amdgcn_wmma_f32_16x16x32_f16(false, a, false, bN, (short)0, accN, false, false);
    }
    // gates (accR/Z include x-projection + b_ih; accN is recurrent-only)
    float hn8[8];
#pragma unroll
    for (int r = 0; r < 8; ++r) {
      int m = r + 8 * half;
      float rg = 1.0f / (1.0f + __expf(-(accR[r] + br)));
      float zg = 1.0f / (1.0f + __expf(-(accZ[r] + bz)));
      float ng = tanhf(xn8[r] + rg * (accN[r] + bn));
      hn8[r] = (1.0f - zg) * ng + zg * hf32[m][n0 + col];
    }
    __syncthreads();   // all A-fragment reads of step s finished
#pragma unroll
    for (int r = 0; r < 8; ++r) {
      int m = r + 8 * half;
      int hcol = n0 + col;
      hf32[m][hcol] = hn8[r];
      hf16[m][hcol] = (_Float16)hn8[r];
      int L = lrow[r];
      int tout = dir ? ((s < L) ? (L - 1 - s) : s) : s;
      float ov = (tout < L) ? hn8[r] : 0.0f;      // mask past length
      size_t off = ((size_t)tout * 16 + m) * D2 + (size_t)dir * HD + hcol;
      outF32[off] = ov;
      if (outF16) outF16[off] = (_Float16)ov;
    }
    __syncthreads();
  }
}

// ---------------------------------------------------------------------------
// final linear + relu + mask -> out[b][t][2]
// ---------------------------------------------------------------------------
__global__ __launch_bounds__(256) void head_kernel(
    const float* __restrict__ h2, const float* __restrict__ lw,
    const float* __restrict__ lb, const int* __restrict__ tlen,
    float* __restrict__ out)
{
  int idx = blockIdx.x * 256 + threadIdx.x;
  if (idx >= BN * TT * 2) return;
  int j = idx & 1;
  int t = (idx >> 1) % TT;
  int b = (idx >> 1) / TT;
  float acc = lb[j];
  const float* hrow = h2 + ((size_t)t * 16 + b) * D2;
  const float* wrow = lw + (size_t)j * D2;
#pragma unroll 4
  for (int k = 0; k < D2; k += 4)
    acc += hrow[k] * wrow[k] + hrow[k+1] * wrow[k+1]
         + hrow[k+2] * wrow[k+2] + hrow[k+3] * wrow[k+3];
  acc = fmaxf(acc, 0.0f);
  out[((size_t)b * TT + t) * 2 + j] = (t < tlen[b]) ? acc : 0.0f;
}

// ---------------------------------------------------------------------------
extern "C" void kernel_launch(void* const* d_in, const int* in_sizes, int n_in,
                              void* d_out, int out_size, void* d_ws, size_t ws_size,
                              hipStream_t stream)
{
  (void)in_sizes; (void)n_in; (void)out_size; (void)ws_size;
  const float* texts  = (const float*)d_in[0];
  const float* w1     = (const float*)d_in[1];
  const float* w2     = (const float*)d_in[2];
  const int*   tlen   = (const int*)d_in[3];
  const int*   mlen   = (const int*)d_in[4];
  const float* w_ih1f = (const float*)d_in[5];  const float* w_hh1f = (const float*)d_in[6];
  const float* b_ih1f = (const float*)d_in[7];  const float* b_hh1f = (const float*)d_in[8];
  const float* w_ih1b = (const float*)d_in[9];  const float* w_hh1b = (const float*)d_in[10];
  const float* b_ih1b = (const float*)d_in[11]; const float* b_hh1b = (const float*)d_in[12];
  const float* w_ih2f = (const float*)d_in[13]; const float* w_hh2f = (const float*)d_in[14];
  const float* b_ih2f = (const float*)d_in[15]; const float* b_hh2f = (const float*)d_in[16];
  const float* w_ih2b = (const float*)d_in[17]; const float* w_hh2b = (const float*)d_in[18];
  const float* b_ih2b = (const float*)d_in[19]; const float* b_hh2b = (const float*)d_in[20];
  const float* lin_w  = (const float*)d_in[21]; const float* lin_b  = (const float*)d_in[22];

  char* ws = (char*)d_ws;
  size_t off = 0;
  auto take = [&](size_t bytes) -> char* {
    char* p = ws + off;
    off += (bytes + 255) & ~(size_t)255;
    return p;
  };
  _Float16* Xh     = (_Float16*)take((size_t)BN * TT * DINP * 2);
  _Float16* Wih1fH = (_Float16*)take((size_t)NG * DINP * 2);
  _Float16* Wih1bH = (_Float16*)take((size_t)NG * DINP * 2);
  _Float16* Whh1fH = (_Float16*)take((size_t)NG * HD * 2);
  _Float16* Whh1bH = (_Float16*)take((size_t)NG * HD * 2);
  _Float16* Wih2fH = (_Float16*)take((size_t)NG * D2 * 2);
  _Float16* Wih2bH = (_Float16*)take((size_t)NG * D2 * 2);
  _Float16* Whh2fH = (_Float16*)take((size_t)NG * HD * 2);
  _Float16* Whh2bH = (_Float16*)take((size_t)NG * HD * 2);
  float*    xWf    = (float*)take((size_t)BN * TT * NG * 4);
  float*    xWb    = (float*)take((size_t)BN * TT * NG * 4);
  float*    h1     = (float*)take((size_t)BN * TT * D2 * 4);
  _Float16* h1h    = (_Float16*)take((size_t)BN * TT * D2 * 2);
  float*    h2     = (float*)take((size_t)BN * TT * D2 * 4);

  auto cvt = [&](const float* s, _Float16* d, int rows, int sc, int dc) {
    int n = rows * dc;
    cvt_pad_kernel<<<(n + 255) / 256, 256, 0, stream>>>(s, d, rows, sc, dc);
  };
  cvt(w_ih1f, Wih1fH, NG, DIN, DINP);
  cvt(w_ih1b, Wih1bH, NG, DIN, DINP);
  cvt(w_hh1f, Whh1fH, NG, HD, HD);
  cvt(w_hh1b, Whh1bH, NG, HD, HD);
  cvt(w_ih2f, Wih2fH, NG, D2, D2);
  cvt(w_ih2b, Wih2bH, NG, D2, D2);
  cvt(w_hh2f, Whh2fH, NG, HD, HD);
  cvt(w_hh2b, Whh2bH, NG, HD, HD);

  feat_kernel<<<dim3(TT, BN), 256, 0, stream>>>(texts, w1, w2, tlen, mlen, Xh);

  const int Mtiles = TT;                    // one 16-row tile per timestep
  const int blocks = Mtiles * (NG / 16) / 8;  // 8 waves per block

  gemm_xw_kernel<<<blocks, 256, 0, stream>>>(Xh, DINP, Wih1fH, b_ih1f, xWf, DINP / 32, Mtiles);
  gemm_xw_kernel<<<blocks, 256, 0, stream>>>(Xh, DINP, Wih1bH, b_ih1b, xWb, DINP / 32, Mtiles);

  gru_scan_kernel<<<2, 1024, 0, stream>>>(xWf, xWb, Whh1fH, Whh1bH, b_hh1f, b_hh1b,
                                          tlen, h1, h1h);

  gemm_xw_kernel<<<blocks, 256, 0, stream>>>(h1h, D2, Wih2fH, b_ih2f, xWf, D2 / 32, Mtiles);
  gemm_xw_kernel<<<blocks, 256, 0, stream>>>(h1h, D2, Wih2bH, b_ih2b, xWb, D2 / 32, Mtiles);

  gru_scan_kernel<<<2, 1024, 0, stream>>>(xWf, xWb, Whh2fH, Whh2bH, b_hh2f, b_hh2b,
                                          tlen, h2, (_Float16*)nullptr);

  head_kernel<<<(BN * TT * 2 + 255) / 256, 256, 0, stream>>>(h2, lin_w, lin_b, tlen,
                                                             (float*)d_out);
}
